// FusionBlock_1056561954968
// MI455X (gfx1250) — compile-verified, matrix operand order checked by reading, not run on hardware
//
#include <hip/hip_runtime.h>
#include <math.h>

// ---------------------------------------------------------------------------
// Types for CDNA5 WMMA (wave32): V_WMMA_F32_16X16X32_BF16
// ---------------------------------------------------------------------------
typedef __attribute__((ext_vector_type(16))) __bf16 v16bf;
typedef __attribute__((ext_vector_type(8)))  __bf16 bf16x8;
typedef __attribute__((ext_vector_type(8)))  float  v8f;

#define DIM      1024
#define HEADS    16
#define HEAD_DIM 64
#define MLP_HID  4096
#define SEQ      1024
#define BATCH    8
#define MTOT     (BATCH * SEQ)   // 8192 rows

static __device__ __forceinline__ v16bf cat8(bf16x8 lo, bf16x8 hi) {
  return __builtin_shufflevector(lo, hi, 0, 1, 2, 3, 4, 5, 6, 7,
                                 8, 9, 10, 11, 12, 13, 14, 15);
}

// LDS byte offset of a shared-memory pointer (low 32 bits of the flat addr).
static __device__ __forceinline__ unsigned lds_off(const void* p) {
  return (unsigned)(unsigned long long)p;
}

// CDNA5 async DMA: global -> LDS, 16 bytes per lane, tracked by ASYNCcnt.
static __device__ __forceinline__ void async_ld16(unsigned lds,
                                                  unsigned long long gaddr) {
  asm volatile("global_load_async_to_lds_b128 %0, %1, off"
               :: "v"(lds), "v"(gaddr) : "memory");
}

// ---------------------------------------------------------------------------
// fp32 -> bf16 conversion (weights)
// ---------------------------------------------------------------------------
__global__ void f32_to_bf16(const float* __restrict__ in, __bf16* __restrict__ out,
                            long long n) {
  long long i = (long long)blockIdx.x * blockDim.x + threadIdx.x;
  long long stride = (long long)gridDim.x * blockDim.x;
  for (; i < n; i += stride) out[i] = (__bf16)in[i];
}

// ---------------------------------------------------------------------------
// LayerNorm over last dim (1024), emits bf16. One 256-thread block per row.
// ---------------------------------------------------------------------------
__global__ __launch_bounds__(256) void layernorm_to_bf16(
    const float* __restrict__ x, const float* __restrict__ g,
    const float* __restrict__ bta, __bf16* __restrict__ out) {
  __shared__ float red[8];
  const long long row = blockIdx.x;
  const float* xr = x + row * DIM;
  const int tid = threadIdx.x;

  float4 xv = *(const float4*)&xr[tid * 4];
  float xe[4] = {xv.x, xv.y, xv.z, xv.w};

  float s = xe[0] + xe[1] + xe[2] + xe[3];
  #pragma unroll
  for (int off = 16; off; off >>= 1) s += __shfl_xor(s, off, 32);
  if ((tid & 31) == 0) red[tid >> 5] = s;
  __syncthreads();
  float mean = 0.f;
  #pragma unroll
  for (int i = 0; i < 8; ++i) mean += red[i];
  mean *= (1.0f / DIM);
  __syncthreads();

  float d0 = xe[0] - mean, d1 = xe[1] - mean, d2 = xe[2] - mean, d3 = xe[3] - mean;
  float vs = d0 * d0 + d1 * d1 + d2 * d2 + d3 * d3;
  #pragma unroll
  for (int off = 16; off; off >>= 1) vs += __shfl_xor(vs, off, 32);
  if ((tid & 31) == 0) red[tid >> 5] = vs;
  __syncthreads();
  float var = 0.f;
  #pragma unroll
  for (int i = 0; i < 8; ++i) var += red[i];
  var *= (1.0f / DIM);
  const float rstd = rsqrtf(var + 1e-5f);

  #pragma unroll
  for (int e = 0; e < 4; ++e) {
    int c = tid * 4 + e;
    float v = (xe[e] - mean) * rstd * g[c] + bta[c];
    out[row * DIM + c] = (__bf16)v;
  }
}

// ---------------------------------------------------------------------------
// GEMM: Out[M,N] = A[M,K](bf16) * W[N,K](bf16)^T.
// Double-buffered LDS fed by GLOBAL_LOAD_ASYNC_TO_LDS_B128 (ASYNCcnt), so the
// async engine copies tile t+1 while the WMMAs chew tile t. Block tile
// 128x128; 8 waves (4x2), wave tile 32x64 (8 wmma per 32-deep K-step).
// Epilogue MODE: 0 = store bf16; 1 = +bias, GELU, store bf16;
//                2 = +bias, +residual, store f32.
// ---------------------------------------------------------------------------
#define BM 128
#define BN 128
#define BK 32
#define LDT 40   // padded LDS row stride (bf16 elems); 80B keeps 16B alignment

template <int MODE>
__global__ __launch_bounds__(256) void gemm_bf16_wmma(
    const __bf16* __restrict__ A, const __bf16* __restrict__ W,
    int M, int N, int K,
    const float* __restrict__ bias, const float* __restrict__ residual,
    float* __restrict__ out_f32, __bf16* __restrict__ out_bf) {
  __shared__ __align__(16) __bf16 As[2][BM][LDT];
  __shared__ __align__(16) __bf16 Bs[2][BN][LDT];

  const int tid  = threadIdx.x;
  const int lane = tid & 31;
  const int wave = tid >> 5;      // 0..7
  const int wm   = wave & 3;      // wave row (4) -> 32 rows each
  const int wn   = wave >> 2;     // wave col (2) -> 64 cols each
  const int half = lane >> 4;     // K-half select per ISA fragment layout
  const int l16  = lane & 15;
  const int m0 = blockIdx.y * BM;
  const int n0 = blockIdx.x * BN;

  // Fixed staging assignment: thread owns 32 contiguous bytes of one row.
  const int sr = tid >> 1;            // 0..127
  const int sc = (tid & 1) * 16;      // 0 or 16
  unsigned long long ga =
      (unsigned long long)(A + (long long)(m0 + sr) * K + sc);
  unsigned long long gb =
      (unsigned long long)(W + (long long)(n0 + sr) * K + sc);
  const unsigned la[2] = {lds_off(&As[0][sr][sc]), lds_off(&As[1][sr][sc])};
  const unsigned lb[2] = {lds_off(&Bs[0][sr][sc]), lds_off(&Bs[1][sr][sc])};

  auto issue = [&](int buf) {
    async_ld16(la[buf],      ga);
    async_ld16(la[buf] + 16, ga + 16);
    async_ld16(lb[buf],      gb);
    async_ld16(lb[buf] + 16, gb + 16);
    ga += BK * 2;   // advance one K-step (bytes)
    gb += BK * 2;
  };

  v8f acc[2][4] = {};
  const int nt = K / BK;

  issue(0);   // tile 0 in flight

  for (int t = 0; t < nt; ++t) {
    const int buf = t & 1;
    if (t + 1 < nt) {
      issue(buf ^ 1);   // tile t+1 in flight (8 outstanding)
      asm volatile("s_wait_asynccnt 0x4" ::: "memory");  // tile t resident
    } else {
      asm volatile("s_wait_asynccnt 0x0" ::: "memory");
    }
    __syncthreads();    // tile t visible to all waves

    // Fragments from LDS (ISA layout: row = lane%16, halves at k = half*8
    // and 16+half*8).
    v16bf af[2], bf[4];
    #pragma unroll
    for (int i = 0; i < 2; ++i) {
      int m = wm * 32 + i * 16 + l16;
      af[i] = cat8(*(const bf16x8*)&As[buf][m][half * 8],
                   *(const bf16x8*)&As[buf][m][16 + half * 8]);
    }
    #pragma unroll
    for (int j = 0; j < 4; ++j) {
      int n = wn * 64 + j * 16 + l16;
      bf[j] = cat8(*(const bf16x8*)&Bs[buf][n][half * 8],
                   *(const bf16x8*)&Bs[buf][n][16 + half * 8]);
    }
    #pragma unroll
    for (int i = 0; i < 2; ++i)
      #pragma unroll
      for (int j = 0; j < 4; ++j)
        acc[i][j] = __builtin_amdgcn_wmma_f32_16x16x32_bf16(
            false, af[i], false, bf[j], (short)0, acc[i][j], false, false);

    __syncthreads();    // all waves done reading buf before it is re-filled
  }

  // Epilogue. C/D layout: lane holds col l16; VGPR r -> row r + 8*half.
  #pragma unroll
  for (int i = 0; i < 2; ++i) {
    #pragma unroll
    for (int j = 0; j < 4; ++j) {
      int n = n0 + wn * 64 + j * 16 + l16;
      float bv = (MODE != 0) ? bias[n] : 0.f;
      #pragma unroll
      for (int r = 0; r < 8; ++r) {
        int m = m0 + wm * 32 + i * 16 + half * 8 + r;
        float v = acc[i][j][r] + bv;
        long long idx = (long long)m * N + n;
        if (MODE == 1) v = 0.5f * v * (1.0f + erff(v * 0.70710678118654752f));
        if (MODE == 2) {
          out_f32[idx] = v + residual[idx];
        } else {
          out_bf[idx] = (__bf16)v;
        }
      }
    }
  }
}

// ---------------------------------------------------------------------------
// Flash attention. qkv: [8192, 3072] bf16, row = b*1024+n, cols [q|k|v] each
// [h*64+d]. One block = (b, h, 128-query slab); each wave owns 16 query rows.
// K tile staged via async-to-LDS; V transposed through registers.
// ---------------------------------------------------------------------------
__global__ __launch_bounds__(256) void attn_wmma(
    const __bf16* __restrict__ qkv, const int* __restrict__ amask,
    __bf16* __restrict__ out) {
  __shared__ __align__(16) __bf16 Kt[32][72];      // 32 keys x 64 d (padded)
  __shared__ __align__(16) __bf16 Vt[64][40];      // transposed: d x 32 keys
  __shared__ __align__(16) __bf16 Pb[8][16][40];   // per-wave P tile 16x32

  const int blk = blockIdx.x;
  const int qb = blk & 7;
  const int h  = (blk >> 3) & 15;
  const int b  = blk >> 7;

  const int tid = threadIdx.x, lane = tid & 31, wave = tid >> 5;
  const int half = lane >> 4, l16 = lane & 15;
  const int q0 = qb * 128 + wave * 16;
  const long long rowbase = (long long)b * SEQ;
  const float scale = 0.125f;  // 1/sqrt(64)

  // Q fragments: 16x64 = two 16x32 k-steps, loaded once.
  v16bf aq[2];
  #pragma unroll
  for (int s = 0; s < 2; ++s) {
    const __bf16* qp = qkv + (rowbase + q0 + l16) * 3072 + h * 64 + s * 32;
    aq[s] = cat8(*(const bf16x8*)(qp + half * 8),
                 *(const bf16x8*)(qp + 16 + half * 8));
  }

  float mrow[8], lrow[8];
  v8f oacc[4] = {};
  #pragma unroll
  for (int r = 0; r < 8; ++r) { mrow[r] = -INFINITY; lrow[r] = 0.f; }

  const int key = tid >> 3;   // 0..31
  const int cb  = tid & 7;    // d chunk
  const unsigned kt_lds = lds_off(&Kt[key][cb * 8]);

  for (int kb = 0; kb < SEQ; kb += 32) {
    __syncthreads();  // previous iteration done with Kt/Vt
    {
      const __bf16* kp =
          qkv + (rowbase + kb + key) * 3072 + 1024 + h * 64 + cb * 8;
      async_ld16(kt_lds, (unsigned long long)kp);
      const __bf16* vp =
          qkv + (rowbase + kb + key) * 3072 + 2048 + h * 64 + cb * 8;
      bf16x8 vv = *(const bf16x8*)vp;
      #pragma unroll
      for (int e = 0; e < 8; ++e) Vt[cb * 8 + e][key] = vv[e];
      asm volatile("s_wait_asynccnt 0x0" ::: "memory");
    }
    __syncthreads();

    // S = Q K^T for two 16-key subtiles (k = head_dim = 64 -> 2 wmma each).
    v8f sacc[2] = {};
    #pragma unroll
    for (int sub = 0; sub < 2; ++sub) {
      int n = sub * 16 + l16;
      #pragma unroll
      for (int s = 0; s < 2; ++s) {
        v16bf bk = cat8(*(const bf16x8*)&Kt[n][s * 32 + half * 8],
                        *(const bf16x8*)&Kt[n][s * 32 + 16 + half * 8]);
        sacc[sub] = __builtin_amdgcn_wmma_f32_16x16x32_bf16(
            false, aq[s], false, bk, (short)0, sacc[sub], false, false);
      }
    }

    float madd0 = (amask[b * SEQ + kb + l16] == 0)      ? -1e30f : 0.f;
    float madd1 = (amask[b * SEQ + kb + 16 + l16] == 0) ? -1e30f : 0.f;

    // Online softmax; row r lives in VGPR slot r across one 16-lane group.
    #pragma unroll
    for (int r = 0; r < 8; ++r) {
      float s0 = sacc[0][r] * scale + madd0;
      float s1 = sacc[1][r] * scale + madd1;
      float rm = fmaxf(s0, s1);
      #pragma unroll
      for (int off = 8; off; off >>= 1) rm = fmaxf(rm, __shfl_xor(rm, off, 32));
      float mnew = fmaxf(mrow[r], rm);
      float corr = __expf(mrow[r] - mnew);
      float p0 = __expf(s0 - mnew);
      float p1 = __expf(s1 - mnew);
      float rs = p0 + p1;
      #pragma unroll
      for (int off = 8; off; off >>= 1) rs += __shfl_xor(rs, off, 32);
      lrow[r] = lrow[r] * corr + rs;
      mrow[r] = mnew;
      #pragma unroll
      for (int dt = 0; dt < 4; ++dt) oacc[dt][r] *= corr;
      int m = half * 8 + r;
      Pb[wave][m][l16]      = (__bf16)p0;
      Pb[wave][m][16 + l16] = (__bf16)p1;
    }
    __syncthreads();  // P visible (uniform across all 8 waves)

    // O += P V  (A = P 16x32, B = V via transposed LDS -> contiguous reads)
    v16bf ap = cat8(*(const bf16x8*)&Pb[wave][l16][half * 8],
                    *(const bf16x8*)&Pb[wave][l16][16 + half * 8]);
    #pragma unroll
    for (int dt = 0; dt < 4; ++dt) {
      v16bf bv = cat8(*(const bf16x8*)&Vt[dt * 16 + l16][half * 8],
                      *(const bf16x8*)&Vt[dt * 16 + l16][16 + half * 8]);
      oacc[dt] = __builtin_amdgcn_wmma_f32_16x16x32_bf16(
          false, ap, false, bv, (short)0, oacc[dt], false, false);
    }
  }

  // Write O / l  -> [row, h*64+d] bf16 (already the (B,N,H,Dh)->(B,N,C) layout)
  #pragma unroll
  for (int dt = 0; dt < 4; ++dt) {
    #pragma unroll
    for (int r = 0; r < 8; ++r) {
      int m = q0 + half * 8 + r;
      int d = dt * 16 + l16;
      out[(rowbase + m) * DIM + h * 64 + d] = (__bf16)(oacc[dt][r] / lrow[r]);
    }
  }
}

// ---------------------------------------------------------------------------
// Launch
// ---------------------------------------------------------------------------
extern "C" void kernel_launch(void* const* d_in, const int* in_sizes, int n_in,
                              void* d_out, int out_size, void* d_ws,
                              size_t ws_size, hipStream_t stream) {
  (void)in_sizes; (void)n_in; (void)out_size; (void)ws_size;
  const float* x      = (const float*)d_in[0];
  const int*   amask  = (const int*)d_in[1];
  const float* ln1_g  = (const float*)d_in[2];
  const float* ln1_b  = (const float*)d_in[3];
  const float* qkv_w  = (const float*)d_in[4];
  const float* proj_w = (const float*)d_in[5];
  const float* proj_b = (const float*)d_in[6];
  const float* ln2_g  = (const float*)d_in[7];
  const float* ln2_b  = (const float*)d_in[8];
  const float* fc1_w  = (const float*)d_in[9];
  const float* fc1_b  = (const float*)d_in[10];
  const float* fc2_w  = (const float*)d_in[11];
  const float* fc2_b  = (const float*)d_in[12];
  float* out = (float*)d_out;

  char* ws = (char*)d_ws;
  size_t off = 0;
  auto take = [&](size_t bytes) -> void* {
    void* p = ws + off;
    off += (bytes + 255) & ~(size_t)255;
    return p;
  };

  __bf16* h1    = (__bf16*)take((size_t)MTOT * DIM * 2);
  __bf16* qkvw  = (__bf16*)take((size_t)3 * DIM * DIM * 2);
  __bf16* projw = (__bf16*)take((size_t)DIM * DIM * 2);
  __bf16* fc1w  = (__bf16*)take((size_t)MLP_HID * DIM * 2);
  __bf16* fc2w  = (__bf16*)take((size_t)DIM * MLP_HID * 2);
  __bf16* qkvb  = (__bf16*)take((size_t)MTOT * 3 * DIM * 2);
  __bf16* attnb = (__bf16*)take((size_t)MTOT * DIM * 2);
  float*  x1    = (float*)take((size_t)MTOT * DIM * 4);
  __bf16* h2    = (__bf16*)take((size_t)MTOT * DIM * 2);
  __bf16* h3    = (__bf16*)take((size_t)MTOT * MLP_HID * 2);

  // Weight conversion fp32 -> bf16
  f32_to_bf16<<<4096, 256, 0, stream>>>(qkv_w,  qkvw,  (long long)3 * DIM * DIM);
  f32_to_bf16<<<4096, 256, 0, stream>>>(proj_w, projw, (long long)DIM * DIM);
  f32_to_bf16<<<4096, 256, 0, stream>>>(fc1_w,  fc1w,  (long long)MLP_HID * DIM);
  f32_to_bf16<<<4096, 256, 0, stream>>>(fc2_w,  fc2w,  (long long)DIM * MLP_HID);

  // --- attention branch ---
  layernorm_to_bf16<<<MTOT, 256, 0, stream>>>(x, ln1_g, ln1_b, h1);
  gemm_bf16_wmma<0><<<dim3(3 * DIM / BN, MTOT / BM), 256, 0, stream>>>(
      h1, qkvw, MTOT, 3 * DIM, DIM, nullptr, nullptr, nullptr, qkvb);
  attn_wmma<<<BATCH * HEADS * (SEQ / 128), 256, 0, stream>>>(qkvb, amask, attnb);
  gemm_bf16_wmma<2><<<dim3(DIM / BN, MTOT / BM), 256, 0, stream>>>(
      attnb, projw, MTOT, DIM, DIM, proj_b, x, x1, nullptr);

  // --- MLP branch ---
  layernorm_to_bf16<<<MTOT, 256, 0, stream>>>(x1, ln2_g, ln2_b, h2);
  gemm_bf16_wmma<1><<<dim3(MLP_HID / BN, MTOT / BM), 256, 0, stream>>>(
      h2, fc1w, MTOT, MLP_HID, DIM, fc1_b, nullptr, nullptr, h3);
  gemm_bf16_wmma<2><<<dim3(DIM / BN, MTOT / BM), 256, 0, stream>>>(
      h3, fc2w, MTOT, DIM, MLP_HID, fc2_b, x1, out, nullptr);
}
